// MultiHeadAttention_30829275250815
// MI455X (gfx1250) — matrix operand order repaired and checked by reading, not compiled, not run on hardware
//
#include <hip/hip_runtime.h>
#include <hip/hip_bf16.h>

// MHA: B=4, T=2048, C=768, H=12, D=64. bf16 WMMA matmuls, fp32 softmax/accum.

#define Bc 4
#define Tc 2048
#define Cc 768
#define Hc 12
#define Dc 64
#define MT (Bc * Tc)   // 8192 tokens

typedef __attribute__((ext_vector_type(16))) __bf16         v16bf;
typedef __attribute__((ext_vector_type(8)))  float          v8f;
typedef __attribute__((ext_vector_type(16))) unsigned short us16;
typedef __attribute__((ext_vector_type(8)))  unsigned short us8;

// ---- helpers -------------------------------------------------------------

__device__ __forceinline__ unsigned short f2bf(float f) {
    union { float f; unsigned u; } v; v.f = f;
    unsigned u = v.u;
    unsigned r = u + 0x7fffu + ((u >> 16) & 1u);   // round-to-nearest-even
    return (unsigned short)(r >> 16);
}

// 16 contiguous bf16 (32B, aligned) -> B-operand fragment
__device__ __forceinline__ v16bf load16(const unsigned short* p) {
    union { us16 u; v16bf b; } cv;
    cv.u = *reinterpret_cast<const us16*>(p);
    return cv.b;
}

// two 8-element (16B) chunks -> A-operand fragment (elems 0..7 / 8..15)
__device__ __forceinline__ v16bf combine8(const unsigned short* plo,
                                          const unsigned short* phi) {
    us8 lo = *reinterpret_cast<const us8*>(plo);
    us8 hi = *reinterpret_cast<const us8*>(phi);
    union { us16 u; v16bf b; } cv;
#pragma unroll
    for (int i = 0; i < 8; ++i) { cv.u[i] = lo[i]; cv.u[i + 8] = hi[i]; }
    return cv.b;
}

__device__ __forceinline__ v8f wmma_bf16(v16bf a, v16bf b, v8f c) {
    // D(16x16,f32) = A(16x32,bf16) x B(32x16,bf16) + C
    return __builtin_amdgcn_wmma_f32_16x16x32_bf16(
        false, a, false, b, (short)0, c, false, false);
}

// ---- conversion kernels --------------------------------------------------

__global__ void k_cvt_x(const float* __restrict__ x,
                        unsigned short* __restrict__ xb, int n) {
    int i = blockIdx.x * 256 + threadIdx.x;
    if (i < n) xb[i] = f2bf(x[i]);
}

// W[k][n] (CxC fp32) -> Wt[n][k] (bf16, N-major) for all four weights
__global__ void k_cvt_w(const float* __restrict__ Wq, const float* __restrict__ Wk,
                        const float* __restrict__ Wv, const float* __restrict__ Wp,
                        unsigned short* __restrict__ Wqt, unsigned short* __restrict__ Wkt,
                        unsigned short* __restrict__ Wvt, unsigned short* __restrict__ Wpt) {
    int i = blockIdx.x * 256 + threadIdx.x;          // 4*768*768 exact
    int which = i / (Cc * Cc);
    int rem   = i % (Cc * Cc);
    int k = rem / Cc, n = rem % Cc;
    const float* W = (which == 0) ? Wq : (which == 1) ? Wk : (which == 2) ? Wv : Wp;
    unsigned short* Wt = (which == 0) ? Wqt : (which == 1) ? Wkt : (which == 2) ? Wvt : Wpt;
    Wt[(size_t)n * Cc + k] = f2bf(W[(size_t)k * Cc + n]);
}

// ---- QKV projection GEMM -------------------------------------------------
// grid.x: M/64 (=128), grid.y: 36 (which*12 + ntile). block = 128 (4 waves).
// Each wave: 16 rows x 64 cols, K-loop over 768 in steps of 32.

__global__ void k_qkv(const unsigned short* __restrict__ xb,
                      const unsigned short* __restrict__ Wqt,
                      const unsigned short* __restrict__ Wkt,
                      const unsigned short* __restrict__ Wvt,
                      unsigned short* __restrict__ Q,
                      unsigned short* __restrict__ K,
                      unsigned short* __restrict__ Vt) {
    const int w    = threadIdx.x >> 5;
    const int lane = threadIdx.x & 31;
    const int half = lane >> 4;
    const int l16  = lane & 15;

    const int m0    = blockIdx.x * 64 + w * 16;
    const int which = blockIdx.y / 12;
    const int n0    = (blockIdx.y % 12) * 64;
    const unsigned short* Wt = (which == 0) ? Wqt : (which == 1) ? Wkt : Wvt;

    const int kbA = half ? 16 : 0;   // A-frag per-lane K offsets {kbA, kbA+8}
    const int kbB = half ? 16 : 0;   // B-frag per-lane K base

    v8f acc[4];
#pragma unroll
    for (int t = 0; t < 4; ++t)
#pragma unroll
        for (int r = 0; r < 8; ++r) acc[t][r] = 0.f;

    const size_t arow = (size_t)(m0 + l16) * Cc;
    for (int kk = 0; kk < Cc; kk += 32) {
        v16bf a = combine8(xb + arow + kk + kbA, xb + arow + kk + kbA + 8);
#pragma unroll
        for (int t = 0; t < 4; ++t) {
            int col = n0 + 16 * t + l16;
            v16bf b = load16(Wt + (size_t)col * Cc + kk + kbB);
            acc[t] = wmma_bf16(a, b, acc[t]);
        }
    }

#pragma unroll
    for (int t = 0; t < 4; ++t) {
#pragma unroll
        for (int r = 0; r < 8; ++r) {
            int m   = m0 + r + 8 * half;           // token index
            int col = n0 + 16 * t + l16;           // channel
            int b   = m / Tc, tok = m % Tc;
            int h   = col / Dc, d = col % Dc;
            float v = acc[t][r];
            size_t bhd = (size_t)(b * Hc + h);
            if (which == 0)
                Q[(bhd * Tc + tok) * Dc + d] = f2bf(v * 0.125f);  // fold D^-0.5
            else if (which == 1)
                K[(bhd * Tc + tok) * Dc + d] = f2bf(v);
            else
                Vt[(bhd * Dc + d) * Tc + tok] = f2bf(v);          // transposed V
        }
    }
}

// ---- flash attention -----------------------------------------------------
// grid.x: T/64 (=32) query blocks, grid.y: B*H (=48). block = 128 (4 waves).
// Wave w handles rows [q0+16w, q0+16w+16), full D=64 output, online softmax.

__global__ void k_attn(const unsigned short* __restrict__ Q,
                       const unsigned short* __restrict__ K,
                       const unsigned short* __restrict__ Vt,
                       unsigned short* __restrict__ O) {
    __shared__ unsigned short Pl[4][16 * 32];      // per-wave P relayout buffer

    const int w    = threadIdx.x >> 5;
    const int lane = threadIdx.x & 31;
    const int half = lane >> 4;
    const int l16  = lane & 15;

    const int bh = blockIdx.y;
    const int b = bh / Hc, h = bh % Hc;
    const int q0 = blockIdx.x * 64;
    const int qw = q0 + w * 16;

    const unsigned short* Qp = Q  + (size_t)(b * Hc + h) * Tc * Dc;
    const unsigned short* Kp = K  + (size_t)(b * Hc + h) * Tc * Dc;
    const unsigned short* Vp = Vt + (size_t)(b * Hc + h) * Dc * Tc;

    const int kbA = half ? 16 : 0;
    const int kbB = half ? 16 : 0;

    // Q A-fragments for D 0..31 and 32..63 (rows fixed for whole loop)
    const size_t qrow = (size_t)(qw + l16) * Dc;
    v16bf aq0 = combine8(Qp + qrow + kbA,      Qp + qrow + 8 + kbA);
    v16bf aq1 = combine8(Qp + qrow + 32 + kbA, Qp + qrow + 40 + kbA);

    float mrow[8], lrow[8];
    v8f acc[4];
#pragma unroll
    for (int r = 0; r < 8; ++r) { mrow[r] = -1e30f; lrow[r] = 0.f; }
#pragma unroll
    for (int t = 0; t < 4; ++t)
#pragma unroll
        for (int r = 0; r < 8; ++r) acc[t][r] = 0.f;

    const int jend = q0 + 64;                      // uniform causal bound
    for (int j = 0; j < jend; j += 32) {
        // ---- scores: two 16x16 tiles, K-dim = D (2 WMMA steps each) ----
        v8f s[2];
#pragma unroll
        for (int t2 = 0; t2 < 2; ++t2) {
            int key = j + 16 * t2 + l16;
            v16bf bk0 = load16(Kp + (size_t)key * Dc + kbB);
            v16bf bk1 = load16(Kp + (size_t)key * Dc + 32 + kbB);
            v8f z;
#pragma unroll
            for (int r = 0; r < 8; ++r) z[r] = 0.f;
            z = wmma_bf16(aq0, bk0, z);
            s[t2] = wmma_bf16(aq1, bk1, z);
        }

        // ---- causal mask + online softmax (fp32) ----
#pragma unroll
        for (int r = 0; r < 8; ++r) {
            int qi = qw + r + 8 * half;
#pragma unroll
            for (int t2 = 0; t2 < 2; ++t2) {
                int ki = j + 16 * t2 + l16;
                if (ki > qi) s[t2][r] = -1e30f;
            }
            float v = fmaxf(s[0][r], s[1][r]);
#pragma unroll
            for (int m = 1; m < 16; m <<= 1) v = fmaxf(v, __shfl_xor(v, m, 16));
            float mnew = fmaxf(mrow[r], v);
            float corr = __expf(mrow[r] - mnew);
            mrow[r] = mnew;
            s[0][r] = __expf(s[0][r] - mnew);
            s[1][r] = __expf(s[1][r] - mnew);
            float rs = s[0][r] + s[1][r];
#pragma unroll
            for (int m = 1; m < 16; m <<= 1) rs += __shfl_xor(rs, m, 16);
            lrow[r] = lrow[r] * corr + rs;
#pragma unroll
            for (int t = 0; t < 4; ++t) acc[t][r] *= corr;
        }

        // ---- P: C-layout -> A-layout via per-wave LDS ----
#pragma unroll
        for (int t2 = 0; t2 < 2; ++t2)
#pragma unroll
            for (int r = 0; r < 8; ++r)
                Pl[w][(r + 8 * half) * 32 + 16 * t2 + l16] = f2bf(s[t2][r]);
        asm volatile("s_wait_dscnt 0x0" ::: "memory");   // wave-local LDS RAW
        v16bf pa = combine8(&Pl[w][l16 * 32 + kbA], &Pl[w][l16 * 32 + 8 + kbA]);

        // ---- O += P x V  (4 N-tiles of 16 over D) ----
#pragma unroll
        for (int t = 0; t < 4; ++t) {
            int drow = 16 * t + l16;
            v16bf bv = load16(Vp + (size_t)drow * Tc + j + kbB);
            acc[t] = wmma_bf16(pa, bv, acc[t]);
        }
    }

    // ---- epilogue: normalize, store bf16 [B,T,C] ----
#pragma unroll
    for (int r = 0; r < 8; ++r) {
        float inv = 1.0f / lrow[r];
        int tok = qw + r + 8 * half;
#pragma unroll
        for (int t = 0; t < 4; ++t) {
            int d = 16 * t + l16;
            O[(size_t)(b * Tc + tok) * Cc + h * Dc + d] = f2bf(acc[t][r] * inv);
        }
    }
}

// ---- output projection GEMM + bias --------------------------------------
// grid.x: M/64 (=128), grid.y: 12 N-tiles. block = 128 (4 waves).

__global__ void k_proj(const unsigned short* __restrict__ Ob,
                       const unsigned short* __restrict__ Wpt,
                       const float* __restrict__ bp,
                       float* __restrict__ out) {
    const int w    = threadIdx.x >> 5;
    const int lane = threadIdx.x & 31;
    const int half = lane >> 4;
    const int l16  = lane & 15;

    const int m0 = blockIdx.x * 64 + w * 16;
    const int n0 = blockIdx.y * 64;
    const int kbA = half ? 16 : 0;
    const int kbB = half ? 16 : 0;

    v8f acc[4];
#pragma unroll
    for (int t = 0; t < 4; ++t)
#pragma unroll
        for (int r = 0; r < 8; ++r) acc[t][r] = 0.f;

    const size_t arow = (size_t)(m0 + l16) * Cc;
    for (int kk = 0; kk < Cc; kk += 32) {
        v16bf a = combine8(Ob + arow + kk + kbA, Ob + arow + kk + kbA + 8);
#pragma unroll
        for (int t = 0; t < 4; ++t) {
            int col = n0 + 16 * t + l16;
            v16bf b = load16(Wpt + (size_t)col * Cc + kk + kbB);
            acc[t] = wmma_bf16(a, b, acc[t]);
        }
    }

#pragma unroll
    for (int t = 0; t < 4; ++t) {
        int col = n0 + 16 * t + l16;
        float bias = bp[col];
#pragma unroll
        for (int r = 0; r < 8; ++r) {
            int m = m0 + r + 8 * half;
            out[(size_t)m * Cc + col] = acc[t][r] + bias;
        }
    }
}

// ---- host launch ---------------------------------------------------------

extern "C" void kernel_launch(void* const* d_in, const int* in_sizes, int n_in,
                              void* d_out, int out_size, void* d_ws, size_t ws_size,
                              hipStream_t stream) {
    const float* x  = (const float*)d_in[0];
    const float* Wq = (const float*)d_in[1];
    const float* Wk = (const float*)d_in[2];
    const float* Wv = (const float*)d_in[3];
    const float* Wp = (const float*)d_in[4];
    const float* bp = (const float*)d_in[5];
    float* out = (float*)d_out;

    char* ws = (char*)d_ws;
    size_t off = 0;
    auto take = [&](size_t bytes) -> char* {
        char* p = ws + off;
        off += (bytes + 255) & ~(size_t)255;
        return p;
    };
    const size_t tokbytes = (size_t)MT * Cc * 2;   // bf16 [8192, 768]
    unsigned short* xb  = (unsigned short*)take(tokbytes);
    unsigned short* Wqt = (unsigned short*)take((size_t)Cc * Cc * 2);
    unsigned short* Wkt = (unsigned short*)take((size_t)Cc * Cc * 2);
    unsigned short* Wvt = (unsigned short*)take((size_t)Cc * Cc * 2);
    unsigned short* Wpt = (unsigned short*)take((size_t)Cc * Cc * 2);
    unsigned short* Qb  = (unsigned short*)take(tokbytes);   // [B,H,T,D]
    unsigned short* Kb  = (unsigned short*)take(tokbytes);   // [B,H,T,D]
    unsigned short* Vtb = (unsigned short*)take(tokbytes);   // [B,H,D,T]
    unsigned short* Ob  = (unsigned short*)take(tokbytes);   // [B,T,C]

    k_cvt_x<<<(MT * Cc) / 256, 256, 0, stream>>>(x, xb, MT * Cc);
    k_cvt_w<<<(4 * Cc * Cc) / 256, 256, 0, stream>>>(Wq, Wk, Wv, Wp,
                                                     Wqt, Wkt, Wvt, Wpt);
    dim3 g1(MT / 64, 36);
    k_qkv<<<g1, 128, 0, stream>>>(xb, Wqt, Wkt, Wvt, Qb, Kb, Vtb);
    dim3 g2(Tc / 64, Bc * Hc);
    k_attn<<<g2, 128, 0, stream>>>(Qb, Kb, Vtb, Ob);
    dim3 g3(MT / 64, Cc / 64);
    k_proj<<<g3, 128, 0, stream>>>(Ob, Wpt, bp, out);
}
